// TensorProductConvLayer_23287312679457
// MI455X (gfx1250) — matrix operand order repaired and checked by reading, not compiled, not run on hardware
//
#include <hip/hip_runtime.h>

// ---------------------------------------------------------------------------
// Fused TensorProductConvLayer for MI455X (gfx1250, wave32, WMMA).
//   w  = relu(edge_attr @ W1 + b1) @ W2 + b2        (bf16 WMMA, f32 accum)
//   tp = FullyConnectedTensorProduct(x[src], sh; w) (register u-reduction)
//   out = segment_mean(tp, dst) + x                 (f32 atomics)
// Never materializes the 655MB per-edge weight tensor in HBM.
// ---------------------------------------------------------------------------

typedef __attribute__((ext_vector_type(8)))  __bf16 v8bf;
typedef __attribute__((ext_vector_type(16))) __bf16 v16bf;
typedef __attribute__((ext_vector_type(8)))  float  v8f;

#define N_NODES   10000
#define N_EDGES   160000
#define EDGE_FDIM 128
#define H_DIM     128
#define WNUM      1024
#define EPB       32      // edges per block (2 WMMA M-tiles)

__device__ __forceinline__ __bf16 tobf(float f) { return (__bf16)f; }

// Swizzled halfword index for a 32x128 bf16 matrix stored as WMMA A-fragments
// (V_WMMA_F32_16X16X32_BF16, 16x32 A tile; ISA 7.12.2 "16-bit A-Matrix 16x32"):
//   lanes 0-15: M=lane, K = Kbase0 + {0..7,16..23}; lanes 16-31: Kbase = 8.
// Each lane's 16 halfs are stored contiguously -> fragment load = 2x ds_load_b128.
__device__ __forceinline__ int a_swz(int m, int k) {
  int mt = m >> 4, mr = m & 15;
  int kt = k >> 5, kk = k & 31;
  int h  = (kk >> 3) & 1;                       // lane-half holding this k
  int j  = (kk & 7) + (((kk >> 4) & 1) << 3);   // half index within lane
  return (((mt * 4 + kt) * 32) + (mr + (h << 4))) * 16 + j;
}

__device__ __forceinline__ v16bf frag16(const __bf16* p) {
  const v8bf* q = (const v8bf*)p;
  v8bf lo = q[0], hi = q[1];
  return __builtin_shufflevector(lo, hi, 0,1,2,3,4,5,6,7,8,9,10,11,12,13,14,15);
}

// ---------------------------------------------------------------------------
// Kernel 0: zero output accumulator + per-node edge counter
// ---------------------------------------------------------------------------
__global__ void k_init(float* __restrict__ out, float* __restrict__ cnt) {
  int i = blockIdx.x * 256 + threadIdx.x;
  if (i < N_NODES * 64) out[i] = 0.0f;
  if (i < N_NODES)      cnt[i] = 0.0f;
}

// ---------------------------------------------------------------------------
// Kernel 1: convert a KxN f32 weight into B-fragment-linear bf16 layout.
// Assumed B layout (32x16 bf16 B tile): lane = (n&15) | ((k>>3)&1)<<4,
// half j = (k&7) | ((k>>4)&1)<<3; each fragment = 32 lanes x 16 contiguous halfs (1KB blocks).
// ---------------------------------------------------------------------------
__global__ void k_swz(const float* __restrict__ w, unsigned short* __restrict__ ws,
                      int K, int N) {
  int t = blockIdx.x * 256 + threadIdx.x;
  if (t >= K * N) return;
  int k = t / N, n = t % N;
  int nt = n >> 4, kt = k >> 5;
  int lane = (n & 15) | (((k >> 3) & 1) << 4);
  int j = (k & 7) | (((k >> 4) & 1) << 3);
  __bf16 b = tobf(w[t]);
  ws[(((nt * (K >> 5)) + kt) * 32 + lane) * 16 + j] =
      __builtin_bit_cast(unsigned short, b);
}

// ---------------------------------------------------------------------------
// Kernel 2: fused edge pipeline. 32 edges / block, 8 waves.
// ---------------------------------------------------------------------------
__global__ __launch_bounds__(256) void k_main(
    const float* __restrict__ x,   const int* __restrict__ ei,
    const float* __restrict__ ea,  const float* __restrict__ esh,
    const unsigned short* __restrict__ w1s_, const float* __restrict__ b1,
    const unsigned short* __restrict__ w2s_, const float* __restrict__ b2,
    float* __restrict__ segout, float* __restrict__ cnt) {

  const __bf16* W1 = (const __bf16*)w1s_;
  const __bf16* W2 = (const __bf16*)w2s_;

  __shared__ __align__(16) __bf16 sA[4096];      // edge_attr tile, A-frag layout
  __shared__ __align__(16) __bf16 sH[4096];      // hidden tile,    A-frag layout
  __shared__ float sOut[EPB][64];                // per-edge tp output accumulator
  __shared__ float sC0[EPB][16];                 // x0*sh0        (path 0)
  __shared__ float sX0[EPB][16];                 // x0            (path 1)
  __shared__ float sC3[EPB][16];                 // dot/sqrt(3)   (path 3)
  __shared__ float sC2[EPB][16][3];              // x1*sh0        (path 2)
  __shared__ float sSh1[EPB][3];

  const int t  = threadIdx.x;
  const int wv = t >> 5;        // wave id 0..7
  const int l  = t & 31;        // lane
  const int nl = l & 15;        // N-column within tile (C/B layout)
  const int hh = l >> 4;        // lane-half (C rows: m = r + 8*hh)
  const int e0 = blockIdx.x * EPB;

  // ---- prefetch next block's edge_attr tile into cache ----
  if (t < 64 && e0 + EPB < N_EDGES) {
    __builtin_prefetch(ea + (size_t)(e0 + EPB) * EDGE_FDIM + t * 64, 0, 3);
  }

  // ---- zero per-edge output accumulator ----
  float* sOutF = (float*)sOut;
  #pragma unroll
  for (int q = 0; q < 8; ++q) sOutF[t + 256 * q] = 0.0f;

  // ---- stage 1: load edge_attr tile -> bf16 A fragments in LDS ----
  {
    const float4* ea4 = (const float4*)(ea + (size_t)e0 * EDGE_FDIM);
    #pragma unroll
    for (int q = 0; q < 4; ++q) {
      int fi = t + 256 * q;           // 0..1023 float4s (32 rows x 32)
      int el = fi >> 5;
      int c4 = fi & 31;
      float4 v = ea4[el * 32 + c4];
      int kb = c4 << 2;
      sA[a_swz(el, kb + 0)] = tobf(v.x);
      sA[a_swz(el, kb + 1)] = tobf(v.y);
      sA[a_swz(el, kb + 2)] = tobf(v.z);
      sA[a_swz(el, kb + 3)] = tobf(v.w);
    }
  }

  // ---- stage 1b: gather x[src], edge_sh; precompute TP coefficients ----
  {
    int el = t >> 3;                 // edge 0..31
    int j  = t & 7;                  // handles u = 2j, 2j+1
    int eg = e0 + el;
    int src = ei[eg];
    float sh0 = esh[eg * 4 + 0];
    float s1x = esh[eg * 4 + 1], s1y = esh[eg * 4 + 2], s1z = esh[eg * 4 + 3];
    if (j == 0) {
      sSh1[el][0] = s1x; sSh1[el][1] = s1y; sSh1[el][2] = s1z;
      atomicAdd(&cnt[ei[N_EDGES + eg]], 1.0f);
    }
    const float* xr = x + (size_t)src * 64;
    #pragma unroll
    for (int uu = 0; uu < 2; ++uu) {
      int u = j * 2 + uu;
      float x0 = xr[u];
      float xa = xr[16 + u * 3 + 0];
      float xb = xr[16 + u * 3 + 1];
      float xc = xr[16 + u * 3 + 2];
      sX0[el][u] = x0;
      sC0[el][u] = x0 * sh0;
      sC3[el][u] = (xa * s1x + xb * s1y + xc * s1z) * 0.57735026918962576f;
      sC2[el][u][0] = xa * sh0;
      sC2[el][u][1] = xb * sh0;
      sC2[el][u][2] = xc * sh0;
    }
  }
  __syncthreads();

  // ---- stage 2: GEMM1 (32x128 @ 128x128), relu+bias -> sH (A-frag layout) ----
  {
    v16bf Bf[4];
    #pragma unroll
    for (int kt = 0; kt < 4; ++kt)
      Bf[kt] = frag16(W1 + ((wv * 4 + kt) * 32 + l) * 16);
    float b1v = b1[wv * 16 + nl];
    #pragma unroll
    for (int mt = 0; mt < 2; ++mt) {
      v8f acc = {};
      #pragma unroll
      for (int kt = 0; kt < 4; ++kt) {
        v16bf Af = frag16(&sA[((mt * 4 + kt) * 32 + l) * 16]);
        acc = __builtin_amdgcn_wmma_f32_16x16x32_bf16(
            false, Af, false, Bf[kt], (short)0, acc, false, false);
      }
      #pragma unroll
      for (int r = 0; r < 8; ++r) {
        float hval = acc[r] + b1v;
        hval = hval > 0.0f ? hval : 0.0f;
        int m = mt * 16 + r + (hh << 3);
        sH[a_swz(m, wv * 16 + nl)] = tobf(hval);
      }
    }
  }
  __syncthreads();

  // ---- stage 3+4: GEMM2 (32x128 @ 128x1024) fused with tensor product ----
  // Column tile nt encodes (path p = nt>>4, u = nt&15); wave wv owns p = wv>>1
  // and u-half (wv&1), reducing over u in registers across its 8 tiles.
  // M-tile is the OUTER loop so the 4 A fragments (32 VGPRs) are loaded from
  // LDS once per M-tile and reused across all 8 N-tiles (B streams from L2).
  {
    const int p = wv >> 1;

    #pragma unroll
    for (int mt = 0; mt < 2; ++mt) {
      v16bf Af[4];
      #pragma unroll
      for (int kt = 0; kt < 4; ++kt)
        Af[kt] = frag16(&sH[((mt * 4 + kt) * 32 + l) * 16]);

      float accA[8];
      float accB[8][3];
      #pragma unroll
      for (int r = 0; r < 8; ++r) {
        accA[r] = 0.0f;
        accB[r][0] = 0.0f; accB[r][1] = 0.0f; accB[r][2] = 0.0f;
      }

      for (int i = 0; i < 8; ++i) {
        int nt = wv * 8 + i;
        int u  = ((wv & 1) << 3) + i;
        float b2v = b2[nt * 16 + nl];

        v8f acc = {};
        #pragma unroll
        for (int kt = 0; kt < 4; ++kt) {
          v16bf Bf = frag16(W2 + ((nt * 4 + kt) * 32 + l) * 16);
          acc = __builtin_amdgcn_wmma_f32_16x16x32_bf16(
              false, Af[kt], false, Bf, (short)0, acc, false, false);
        }

        #pragma unroll
        for (int r = 0; r < 8; ++r) {
          int m = mt * 16 + r + (hh << 3);
          float wval = acc[r] + b2v;
          if (p == 0) {
            accA[r] += sC0[m][u] * wval;           // 0e x 0e -> 0e
          } else if (p == 1) {
            accA[r] += sX0[m][u] * wval;           // 0e x 1o -> 1o
          } else if (p == 3) {
            accA[r] += sC3[m][u] * wval;           // 1o x 1o -> 0e
          } else {
            accB[r][0] += sC2[m][u][0] * wval;     // 1o x 0e -> 1o
            accB[r][1] += sC2[m][u][1] * wval;
            accB[r][2] += sC2[m][u][2] * wval;
          }
        }
      }

      // flush u-half partials into the per-edge accumulator (ds_add_f32)
      #pragma unroll
      for (int r = 0; r < 8; ++r) {
        int m = mt * 16 + r + (hh << 3);
        if (p == 0 || p == 3) {
          atomicAdd(&sOut[m][nl], accA[r]);
        } else if (p == 1) {
          atomicAdd(&sOut[m][16 + nl * 3 + 0], accA[r] * sSh1[m][0]);
          atomicAdd(&sOut[m][16 + nl * 3 + 1], accA[r] * sSh1[m][1]);
          atomicAdd(&sOut[m][16 + nl * 3 + 2], accA[r] * sSh1[m][2]);
        } else {
          atomicAdd(&sOut[m][16 + nl * 3 + 0], accB[r][0]);
          atomicAdd(&sOut[m][16 + nl * 3 + 1], accB[r][1]);
          atomicAdd(&sOut[m][16 + nl * 3 + 2], accB[r][2]);
        }
      }
    }
  }
  __syncthreads();

  // ---- stage 5: scatter to nodes (global_atomic_add_f32) ----
  {
    const float alpha = 0.17677669529663687f;  // 1/sqrt(2*MUL)
    #pragma unroll
    for (int q = 0; q < 8; ++q) {
      int idx = t + 256 * q;                   // 0..2047
      int el  = idx >> 6;
      int c   = idx & 63;
      int dst = ei[N_EDGES + e0 + el];
      atomicAdd(&segout[(size_t)dst * 64 + c], alpha * sOutF[idx]);
    }
  }
}

// ---------------------------------------------------------------------------
// Kernel 3: segment mean + residual
// ---------------------------------------------------------------------------
__global__ void k_fin(float* __restrict__ out, const float* __restrict__ cnt,
                      const float* __restrict__ x) {
  int i = blockIdx.x * 256 + threadIdx.x;
  if (i >= N_NODES * 64) return;
  float c = cnt[i >> 6];
  out[i] = out[i] / fmaxf(c, 1.0f) + x[i];
}

// ---------------------------------------------------------------------------
extern "C" void kernel_launch(void* const* d_in, const int* in_sizes, int n_in,
                              void* d_out, int out_size, void* d_ws, size_t ws_size,
                              hipStream_t stream) {
  const float* x   = (const float*)d_in[0];
  const int*   ei  = (const int*)d_in[1];
  const float* ea  = (const float*)d_in[2];
  const float* esh = (const float*)d_in[3];
  const float* w1  = (const float*)d_in[4];
  const float* b1  = (const float*)d_in[5];
  const float* w2  = (const float*)d_in[6];
  const float* b2  = (const float*)d_in[7];
  float* out = (float*)d_out;

  char* ws = (char*)d_ws;
  unsigned short* w1s = (unsigned short*)ws;                       // 32 KB
  unsigned short* w2s = (unsigned short*)(ws + 32768);             // 256 KB
  float*          cnt = (float*)(ws + 32768 + 262144);             // 40 KB

  k_init<<<(N_NODES * 64 + 255) / 256, 256, 0, stream>>>(out, cnt);
  k_swz<<<(H_DIM * H_DIM + 255) / 256, 256, 0, stream>>>(w1, w1s, H_DIM, H_DIM);
  k_swz<<<(H_DIM * WNUM + 255) / 256, 256, 0, stream>>>(w2, w2s, H_DIM, WNUM);
  k_main<<<N_EDGES / EPB, 256, 0, stream>>>(x, ei, ea, esh, w1s, b1, w2s, b2,
                                            out, cnt);
  k_fin<<<(N_NODES * 64 + 255) / 256, 256, 0, stream>>>(out, cnt, x);
}
